// GATConv_68642167324741
// MI455X (gfx1250) — matrix operand order repaired and checked by reading, not compiled, not run on hardware
//
#include <hip/hip_runtime.h>
#include <hip/hip_bf16.h>

typedef __attribute__((ext_vector_type(16))) _Float16 v16h;
typedef __attribute__((ext_vector_type(8)))  _Float16 v8h;
typedef __attribute__((ext_vector_type(8)))  float    v8f;

#define ALPHA 0.2f
#define B_    8
#define N_    2048
#define F_    128
#define OUT_  128
#define H_    4
#define D_    32

// ---------------------------------------------------------------------------
// Kernel 0: WT[n][k] = (f16) W[k][n]   (B-fragment friendly: contiguous in k)
// ---------------------------------------------------------------------------
__global__ void k_wt(const float* __restrict__ W, _Float16* __restrict__ WT) {
    int idx = blockIdx.x * 256 + threadIdx.x;      // 128*128 = 16384
    int n = idx & (OUT_ - 1);                      // contiguous reads over n
    int k = idx >> 7;
    WT[(size_t)n * F_ + k] = (_Float16)W[(size_t)k * OUT_ + n];
}

// ---------------------------------------------------------------------------
// Kernel 1: h = x @ W via WMMA f16, write f16 hT[b][head][d][n] (n innermost)
// One wave per 16x16 output tile. M = B*N = 16384 rows, N = OUT = 128 cols.
// ---------------------------------------------------------------------------
__global__ void k_proj(const float* __restrict__ x,
                       const _Float16* __restrict__ WT,
                       _Float16* __restrict__ hT) {
    const int lane = threadIdx.x & 31;
    const int wid  = (blockIdx.x * blockDim.x + threadIdx.x) >> 5;  // 0..8191
    const int tM   = wid >> 3;        // 1024 row tiles
    const int tN   = wid & 7;         // 8 col tiles
    const int r    = lane & 15;
    const int hi   = lane >> 4;
    const int kb   = hi * 8;          // A-fragment K base (per ISA layout)
    const int kb16 = hi * 16;         // unused (B fragment uses kb too)

    const int mG  = tM * 16 + r;      // A row
    const int col = tN * 16 + r;      // B column / C column
    const float*    xrow = x  + (size_t)mG  * F_;
    const _Float16* wrow = WT + (size_t)col * F_;

    v8f acc = {};
#pragma unroll
    for (int kk = 0; kk < F_; kk += 32) {
        // A fragment: lane holds row mG, K = {kk+kb..+7} and {kk+16+kb..+7}
        float4 a0 = *(const float4*)(xrow + kk + kb);
        float4 a1 = *(const float4*)(xrow + kk + kb + 4);
        float4 a2 = *(const float4*)(xrow + kk + 16 + kb);
        float4 a3 = *(const float4*)(xrow + kk + 16 + kb + 4);
        v16h A;
        A[0]=(_Float16)a0.x; A[1]=(_Float16)a0.y; A[2]=(_Float16)a0.z; A[3]=(_Float16)a0.w;
        A[4]=(_Float16)a1.x; A[5]=(_Float16)a1.y; A[6]=(_Float16)a1.z; A[7]=(_Float16)a1.w;
        A[8]=(_Float16)a2.x; A[9]=(_Float16)a2.y; A[10]=(_Float16)a2.z; A[11]=(_Float16)a2.w;
        A[12]=(_Float16)a3.x; A[13]=(_Float16)a3.y; A[14]=(_Float16)a3.z; A[15]=(_Float16)a3.w;
        // B fragment: K = {kk+kb..+7} and {kk+16+kb..+7}, matching A; 16B aligned
        v8h b0 = *(const v8h*)(wrow + kk + kb);
        v8h b1 = *(const v8h*)(wrow + kk + 16 + kb);
        v16h Bf = __builtin_shufflevector(b0, b1, 0,1,2,3,4,5,6,7,8,9,10,11,12,13,14,15);
        acc = __builtin_amdgcn_wmma_f32_16x16x32_f16(false, A, false, Bf,
                                                     (short)0, acc, false, false);
    }

    // C layout: lane holds column `col`; VGPR v -> row tM*16 + 8*hi + v.
    const int head  = col >> 5;
    const int d     = col & 31;
    const int rowTop = tM * 16 + 8 * hi;
    const int bb     = rowTop >> 11;           // 2048 rows per batch, tiles don't cross
    const int nbase  = rowTop & (N_ - 1);      // 8 consecutive nodes
    v8h hv;
#pragma unroll
    for (int v = 0; v < 8; ++v) hv[v] = (_Float16)acc[v];
    _Float16* dst = hT + ((((size_t)bb * H_ + head) * D_ + d) * N_ + nbase);
    *(v8h*)dst = hv;   // 16B aligned (nbase multiple of 8)
}

// ---------------------------------------------------------------------------
// Kernel 2: eS[b][h][n] = sum_d hT[b][h][d][n]*a_src[h][d]; same for eD.
// ---------------------------------------------------------------------------
__global__ void k_edges(const _Float16* __restrict__ hT,
                        const float* __restrict__ a_src,
                        const float* __restrict__ a_dst,
                        float* __restrict__ eS, float* __restrict__ eD) {
    int idx  = blockIdx.x * 256 + threadIdx.x;   // B*H*N = 65536
    int n    = idx & (N_ - 1);
    int bh   = idx >> 11;
    int head = bh & (H_ - 1);
    const _Float16* hp = hT + (size_t)bh * D_ * N_ + n;
    float es = 0.f, ed = 0.f;
#pragma unroll
    for (int d = 0; d < D_; ++d) {
        float hv = (float)hp[(size_t)d * N_];
        es += hv * a_src[head * D_ + d];
        ed += hv * a_dst[head * D_ + d];
    }
    eS[idx] = es;
    eD[idx] = ed;
}

// ---------------------------------------------------------------------------
// Kernel 3: fused masked-softmax attention + aggregation.
// One wave per (b, head, 16-row i tile). Two passes over j in chunks of 32:
// pass 1 = row max, pass 2 = p=exp(e-m) packed into WMMA A-frag, 2 WMMAs/chunk.
// ---------------------------------------------------------------------------
__global__ void k_attn(const float* __restrict__ adj,
                       const float* __restrict__ eS,
                       const float* __restrict__ eD,
                       const _Float16* __restrict__ hT,
                       float* __restrict__ out) {
    const int lane = threadIdx.x & 31;
    const int wid  = (blockIdx.x * blockDim.x + threadIdx.x) >> 5;  // 0..4095
    const int iT   = wid & 127;
    const int bh   = wid >> 7;             // 0..31
    const int head = bh & (H_ - 1);
    const int b    = bh >> 2;
    const int r    = lane & 15;
    const int hi   = lane >> 4;
    const int kb   = hi * 8;               // A-frag K base
    const int kb16 = hi * 16;              // unused (B frag uses kb too)
    const int i    = iT * 16 + r;          // this lane's attention row

    const float     esr    = eS[(size_t)bh * N_ + i];
    const float*    adjrow = adj + (size_t)i  * N_;
    const float*    edrow  = eD  + (size_t)bh * N_;
    const _Float16* hbase  = hT  + (size_t)bh * D_ * N_;

    // ---- pass 1: masked row max ----
    float m = -1e30f;
    for (int c = 0; c < N_; c += 32) {
        const int j0 = c + kb, j1 = c + 16 + kb;
        float4 A0 = *(const float4*)(adjrow + j0);
        float4 A1 = *(const float4*)(adjrow + j0 + 4);
        float4 A2 = *(const float4*)(adjrow + j1);
        float4 A3 = *(const float4*)(adjrow + j1 + 4);
        float4 E0 = *(const float4*)(edrow + j0);
        float4 E1 = *(const float4*)(edrow + j0 + 4);
        float4 E2 = *(const float4*)(edrow + j1);
        float4 E3 = *(const float4*)(edrow + j1 + 4);
        float av[16] = {A0.x,A0.y,A0.z,A0.w, A1.x,A1.y,A1.z,A1.w,
                        A2.x,A2.y,A2.z,A2.w, A3.x,A3.y,A3.z,A3.w};
        float ev[16] = {E0.x,E0.y,E0.z,E0.w, E1.x,E1.y,E1.z,E1.w,
                        E2.x,E2.y,E2.z,E2.w, E3.x,E3.y,E3.z,E3.w};
#pragma unroll
        for (int t = 0; t < 16; ++t) {
            int j = (t < 8) ? (j0 + t) : (j1 + t - 8);
            float e = esr + ev[t];
            e = (e > 0.f) ? e : ALPHA * e;
            bool ok = (av[t] > 0.f) || (j == i);
            if (ok) m = fmaxf(m, e);
        }
    }
    m = fmaxf(m, __shfl_xor(m, 16));

    // ---- pass 2: p = exp(e - m), accumulate l and WMMA aggregation ----
    float l = 0.f;
    v8f c0 = {};   // d columns 0..15
    v8f c1 = {};   // d columns 16..31
    for (int c = 0; c < N_; c += 32) {
        const int j0 = c + kb, j1 = c + 16 + kb;
        float4 A0 = *(const float4*)(adjrow + j0);
        float4 A1 = *(const float4*)(adjrow + j0 + 4);
        float4 A2 = *(const float4*)(adjrow + j1);
        float4 A3 = *(const float4*)(adjrow + j1 + 4);
        float4 E0 = *(const float4*)(edrow + j0);
        float4 E1 = *(const float4*)(edrow + j0 + 4);
        float4 E2 = *(const float4*)(edrow + j1);
        float4 E3 = *(const float4*)(edrow + j1 + 4);
        float av[16] = {A0.x,A0.y,A0.z,A0.w, A1.x,A1.y,A1.z,A1.w,
                        A2.x,A2.y,A2.z,A2.w, A3.x,A3.y,A3.z,A3.w};
        float ev[16] = {E0.x,E0.y,E0.z,E0.w, E1.x,E1.y,E1.z,E1.w,
                        E2.x,E2.y,E2.z,E2.w, E3.x,E3.y,E3.z,E3.w};
        v16h P;
#pragma unroll
        for (int t = 0; t < 16; ++t) {
            int j = (t < 8) ? (j0 + t) : (j1 + t - 8);
            float e = esr + ev[t];
            e = (e > 0.f) ? e : ALPHA * e;
            bool ok = (av[t] > 0.f) || (j == i);
            float p = ok ? __expf(e - m) : 0.f;
            l += p;
            P[t] = (_Float16)p;
        }
        // B fragments: fixed d = column, j = {j0..j0+7} and {j1..j1+7} (16B aligned)
        v8h B0l = *(const v8h*)(hbase + (size_t)r        * N_ + j0);
        v8h B0h = *(const v8h*)(hbase + (size_t)r        * N_ + j1);
        v8h B1l = *(const v8h*)(hbase + (size_t)(r + 16) * N_ + j0);
        v8h B1h = *(const v8h*)(hbase + (size_t)(r + 16) * N_ + j1);
        v16h B0 = __builtin_shufflevector(B0l, B0h, 0,1,2,3,4,5,6,7,8,9,10,11,12,13,14,15);
        v16h B1 = __builtin_shufflevector(B1l, B1h, 0,1,2,3,4,5,6,7,8,9,10,11,12,13,14,15);
        c0 = __builtin_amdgcn_wmma_f32_16x16x32_f16(false, P, false, B0,
                                                    (short)0, c0, false, false);
        c1 = __builtin_amdgcn_wmma_f32_16x16x32_f16(false, P, false, B1,
                                                    (short)0, c1, false, false);
    }
    l += __shfl_xor(l, 16);

    // ---- normalize and store; C layout: lane col = r, VGPR v row = 8*hi+v ----
#pragma unroll
    for (int v = 0; v < 8; ++v) {
        int row = 8 * hi + v;
        float lr = __shfl(l, row);            // lane `row` holds that row's sum
        float inv = 1.0f / lr;
        int iG = iT * 16 + row;
        size_t base = ((size_t)b * N_ + iG) * OUT_ + head * D_;
        out[base + r]      = c0[v] * inv;
        out[base + 16 + r] = c1[v] * inv;
    }
}

// ---------------------------------------------------------------------------
extern "C" void kernel_launch(void* const* d_in, const int* in_sizes, int n_in,
                              void* d_out, int out_size, void* d_ws, size_t ws_size,
                              hipStream_t stream) {
    const float* x     = (const float*)d_in[0];   // [8, 2048, 128]
    const float* adj   = (const float*)d_in[1];   // [2048, 2048]
    const float* W     = (const float*)d_in[2];   // [128, 128]
    const float* a_src = (const float*)d_in[3];   // [4, 32]
    const float* a_dst = (const float*)d_in[4];   // [4, 32]
    float* out = (float*)d_out;                   // [8, 2048, 128]

    char* ws = (char*)d_ws;
    _Float16* hT = (_Float16*)ws;                                   // 4 MiB
    _Float16* WT = (_Float16*)(ws + (size_t)(4 << 20));             // 32 KiB
    float* eS = (float*)(ws + (size_t)(4 << 20) + (32 << 10));      // 256 KiB
    float* eD = eS + (size_t)B_ * H_ * N_;                          // 256 KiB

    // 0) W -> f16 transposed
    k_wt<<<(OUT_ * F_) / 256, 256, 0, stream>>>(W, WT);
    // 1) projection GEMM -> hT  (8192 waves, 8 waves/block)
    k_proj<<<1024, 256, 0, stream>>>(x, WT, hT);
    // 2) attention edge terms
    k_edges<<<(B_ * H_ * N_) / 256, 256, 0, stream>>>(hT, a_src, a_dst, eS, eD);
    // 3) fused masked softmax + aggregation (4096 waves, 8 waves/block)
    k_attn<<<512, 256, 0, stream>>>(adj, eS, eD, hT, out);

    (void)in_sizes; (void)n_in; (void)out_size; (void)ws_size;
}
